// MoA_33724083208855
// MI455X (gfx1250) — matrix operand order repaired and checked by reading, not compiled, not run on hardware
//
#include <hip/hip_runtime.h>

typedef _Float16 f16;
typedef __attribute__((ext_vector_type(16))) _Float16 v16h;
typedef __attribute__((ext_vector_type(8)))  float    v8f;
typedef __attribute__((ext_vector_type(4)))  int      i4;

union AB { v16h v; i4 q[2]; };

// ---------------------------------------------------------------------------
// Sizes
// ---------------------------------------------------------------------------
#define NTOK   16384      // B*T
#define DDIM   1024
#define NEXP   8
#define HDIM   128
#define TOKWG  128        // tokens per workgroup (8 waves x 16)

// workspace byte offsets
#define WS_XC   0ull                    // f16 xC  [32][16384][32]  (chunk-major)  32 MB
#define WS_WDC  33554432ull             // f16 WdC [32][1024][32]                   2 MB
#define WS_WUT  35651584ull             // f16 WuT [1024 n][1024 k]                 2 MB
#define WS_WRT  37748736ull             // f16 WrT [16 n][1024 k] (rows 8..15 = 0) 32 KB
#define WS_BUT  37781504ull             // f16 buT [1024 n][32 k] (k>=8 -> 0)      64 KB
#define WS_END  37847040ull

// ---------------------------------------------------------------------------
// WMMA helpers (CDNA5 v_wmma_f32_16x16x32_f16)
// ---------------------------------------------------------------------------
__device__ __forceinline__ v8f wmma16(const AB& a, const AB& b, v8f c) {
    return __builtin_amdgcn_wmma_f32_16x16x32_f16(false, a.v, false, b.v,
                                                  (short)0, c, false, false);
}

__device__ __forceinline__ v8f vzero8() {
    v8f z;
#pragma unroll
    for (int i = 0; i < 8; ++i) z[i] = 0.0f;
    return z;
}

// A fragment (16x32 f16) from row-major [rows][rowStride] storage.
// lane<16: K runs {0..7, 16..23}; lane>=16: {8..15, 24..31}  (ISA 7.12.2)
__device__ __forceinline__ AB load_A(const f16* base, int rowStride,
                                     int r0, int c0, int m16, int hi) {
    AB a;
    const f16* p = base + (size_t)(r0 + m16) * rowStride + c0 + hi * 8;
    a.q[0] = *(const i4*)p;
    a.q[1] = *(const i4*)(p + 16);
    return a;
}

// B fragment (32x16 f16) from N-major [n][rowStride(=K)] storage.
// lane<16: K runs {0..7, 16..23}; lane>=16: {8..15, 24..31} (matches load_A)
__device__ __forceinline__ AB load_B(const f16* base, int rowStride,
                                     int n0, int c0, int m16, int hi) {
    AB b;
    const f16* p = base + (size_t)(n0 + m16) * rowStride + c0 + hi * 8;
    b.q[0] = *(const i4*)p;
    b.q[1] = *(const i4*)(p + 16);
    return b;
}

// ---------------------------------------------------------------------------
// CDNA5 async global->LDS copy (16B per lane), tracked by ASYNCcnt.
// LDS address = low 32 bits of the flat pointer (ISA 10.2 aperture mapping).
// ---------------------------------------------------------------------------
__device__ __forceinline__ void a_cp16(const f16* g, const f16* l) {
    unsigned            lds = (unsigned)(uintptr_t)l;
    unsigned long long  ga  = (unsigned long long)(uintptr_t)g;
    asm volatile("global_load_async_to_lds_b128 %0, %1, off"
                 :: "v"(lds), "v"(ga) : "memory");
}

__device__ __forceinline__ void wait_async() {
    asm volatile("s_wait_asynccnt 0x0" ::: "memory");
}

// ---------------------------------------------------------------------------
// Prepass kernels: fp32 -> f16, layouts optimized for WMMA fragment loads
// ---------------------------------------------------------------------------
__global__ void prep_x(const float* __restrict__ x, f16* __restrict__ xC) {
    unsigned o = blockIdx.x * 256u + threadIdx.x;          // 16777216 elements
    unsigned kc = o >> 19, rem = o & 0x7FFFFu;
    unsigned t = rem >> 5, ki = rem & 31u;
    xC[o] = (f16)x[(size_t)t * DDIM + (kc << 5) + ki];
}

__global__ void prep_wd(const float* __restrict__ Wd, f16* __restrict__ WdC) {
    unsigned o = blockIdx.x * 256u + threadIdx.x;          // 1048576 elements
    unsigned kc = o >> 15, rem = o & 32767u;
    unsigned n = rem >> 5, ki = rem & 31u;
    unsigned e = n >> 7, h = n & 127u, k = (kc << 5) + ki;
    WdC[o] = (f16)Wd[((size_t)e * DDIM + k) * HDIM + h];
}

__global__ void prep_wu(const float* __restrict__ Wu, f16* __restrict__ WuT) {
    unsigned o = blockIdx.x * 256u + threadIdx.x;          // 1048576 elements
    unsigned n = o >> 10, k = o & 1023u;
    unsigned e = k >> 7, h = k & 127u;
    WuT[o] = (f16)Wu[((size_t)e * HDIM + h) * DDIM + n];
}

__global__ void prep_wr(const float* __restrict__ Wr, f16* __restrict__ WrT) {
    unsigned o = blockIdx.x * 256u + threadIdx.x;          // 16384 elements
    unsigned n = o >> 10, k = o & 1023u;
    WrT[o] = (n < NEXP) ? (f16)Wr[k * NEXP + n] : (f16)0.0f;
}

__global__ void prep_bu(const float* __restrict__ bu, f16* __restrict__ buT) {
    unsigned o = blockIdx.x * 256u + threadIdx.x;          // 32768 elements
    unsigned n = o >> 5, k = o & 31u;
    buT[o] = (k < NEXP) ? (f16)bu[k * DDIM + n] : (f16)0.0f;
}

// ---------------------------------------------------------------------------
// Async staging (issue-only; completion via s_wait_asynccnt at loop top)
// ---------------------------------------------------------------------------
__device__ __forceinline__ void stage_down(f16* dst, const f16* xC, const f16* WdC,
                                           int tok0, int e, int kc, int tid) {
    const f16* xs = xC + ((size_t)kc * NTOK + tok0) * 32;     // 8 KB activations
    const f16* ws = WdC + ((size_t)kc * 1024 + e * 128) * 32; // 8 KB weights
#pragma unroll
    for (int r = 0; r < 2; ++r) {
        int o = (tid + r * 256) * 8;          // halves (16 B per lane)
        a_cp16(xs + o, dst + o);
        a_cp16(ws + o, dst + 4096 + o);
    }
}

__device__ __forceinline__ void stage_up(f16* dst, const f16* WuT, const f16* buT,
                                         int nt, int tid) {
    const f16* ws = WuT + (size_t)nt * 16 * 1024;             // 32 KB weights
#pragma unroll
    for (int r = 0; r < 8; ++r) {
        int o = (tid + r * 256) * 8;
        a_cp16(ws + o, dst + o);
    }
    if (tid < 64) {                                            // 1 KB bias slice
        int o = tid * 8;
        a_cp16(buT + (size_t)nt * 512 + o, dst + 16384 + o);
    }
}

// ---------------------------------------------------------------------------
// Main fused kernel: router softmax + down GEMM + GELU + up GEMM
// 256 threads = 8 waves; wave w owns tokens [blk*128 + w*16, +16)
// ---------------------------------------------------------------------------
__global__ __launch_bounds__(256, 1) void moa_main(
    const f16* __restrict__ xC, const f16* __restrict__ WdC,
    const f16* __restrict__ WuT, const f16* __restrict__ WrT,
    const f16* __restrict__ buT, const float* __restrict__ br,
    const float* __restrict__ bd, float* __restrict__ out)
{
    __shared__ alignas(16) f16 stage[2][16896];  // down: xc[0..4095]+wc[4096..8191]; up: WuT slice + buT slice
    __shared__ alignas(16) f16 gscr[8][2048];    // per-wave: logits / probs / gelu transpose scratch
    __shared__ float pf[8][128];                 // per-wave probs f32 [16 tok][8 exp]
    __shared__ float bds[NEXP * HDIM];
    __shared__ float brs[NEXP];

    const int tid  = threadIdx.x;
    const int wave = tid >> 5, lane = tid & 31;
    const int m16  = lane & 15, hi = lane >> 4;
    const int tok0 = blockIdx.x * TOKWG;
    const int trow0 = wave * 16;

    for (int i = tid; i < NEXP * HDIM; i += 256) bds[i] = bd[i];
    if (tid < NEXP) brs[tid] = br[tid];

    // kick off the first down-phase stage; it streams in under the router GEMM
    stage_down(stage[0], xC, WdC, tok0, 0, 0, tid);
    __syncthreads();   // bds/brs visibility

    // ---------------- router: logits = x @ Wr (WMMA), softmax in LDS ----------
    v8f cr = vzero8();
    for (int kc = 0; kc < 32; ++kc) {
        AB a = load_A(xC + (size_t)kc * NTOK * 32, 32, tok0 + trow0, 0, m16, hi);
        AB b = load_B(WrT, 1024, 0, kc * 32, m16, hi);
        cr = wmma16(a, b, cr);
    }
    float* ls = (float*)&gscr[wave][0];          // [16 m][16 n] f32 scratch
#pragma unroll
    for (int i = 0; i < 8; ++i) ls[(i + hi * 8) * 16 + m16] = cr[i];
    asm volatile("s_wait_dscnt 0" ::: "memory");
    f16* pgs = &gscr[wave][1024];                // probs f16 [16][32] zero-padded
    if (lane < 16) {
        float l[NEXP], mx = -1e30f;
#pragma unroll
        for (int e = 0; e < NEXP; ++e) { l[e] = ls[lane * 16 + e] + brs[e]; mx = fmaxf(mx, l[e]); }
        float s = 0.0f;
#pragma unroll
        for (int e = 0; e < NEXP; ++e) { l[e] = expf(l[e] - mx); s += l[e]; }
        float inv = 1.0f / s;
#pragma unroll
        for (int e = 0; e < NEXP; ++e) {
            float p = l[e] * inv;
            pf[wave][lane * 8 + e] = p;
            pgs[lane * 32 + e] = (f16)p;
        }
#pragma unroll
        for (int e = NEXP; e < 32; ++e) pgs[lane * 32 + e] = (f16)0.0f;
    }
    asm volatile("s_wait_dscnt 0" ::: "memory");
    AB pgf = load_A(pgs, 32, 0, 0, m16, hi);     // probs A-fragment, kept in regs

    // ---------------- down: per-expert 16x128 <- 16x1024 @ 1024x128 -----------
    AB gfrag[32];                                 // prob-scaled gelu activations (A frags)
#pragma unroll
    for (int e = 0; e < NEXP; ++e) {
        v8f acc[8];
#pragma unroll
        for (int j = 0; j < 8; ++j) acc[j] = vzero8();
        for (int kc = 0; kc < 32; ++kc) {
            int step = e * 32 + kc;
            int buf = step & 1;
            // wait for the chunk staged one iteration ago, make it WG-visible
            wait_async();
            __syncthreads();
            // prefetch next chunk into the other buffer (flies during compute)
            if (step + 1 < NEXP * 32) {
                int ns = step + 1;
                stage_down(stage[buf ^ 1], xC, WdC, tok0, ns >> 5, ns & 31, tid);
            }
            AB a = load_A(&stage[buf][0], 32, trow0, 0, m16, hi);
#pragma unroll
            for (int j = 0; j < 8; ++j) {
                AB b = load_B(&stage[buf][4096], 32, j * 16, 0, m16, hi);
                acc[j] = wmma16(a, b, acc[j]);
            }
        }
        // bias + exact GELU + fold router prob, transpose via per-wave LDS
        float pv[8];
#pragma unroll
        for (int i = 0; i < 8; ++i) pv[i] = pf[wave][(i + hi * 8) * 8 + e];
#pragma unroll
        for (int j = 0; j < 8; ++j) {
            float bdv = bds[e * HDIM + j * 16 + m16];
#pragma unroll
            for (int i = 0; i < 8; ++i) {
                float xv = acc[j][i] + bdv;
                float g = 0.5f * xv * (1.0f + erff(xv * 0.70710678118654752f));
                gscr[wave][(i + hi * 8) * 128 + j * 16 + m16] = (f16)(g * pv[i]);
            }
        }
        asm volatile("s_wait_dscnt 0" ::: "memory");
#pragma unroll
        for (int q = 0; q < 4; ++q)
            gfrag[e * 4 + q] = load_A(&gscr[wave][0], 128, 0, q * 32, m16, hi);
    }

    // ---------------- up: out tile = pg@buT + g'@WuT, per 16-col tile ---------
    // stage[0] reads all completed before the last down barrier -> safe to refill
    stage_up(stage[0], WuT, buT, 0, tid);
    for (int nt = 0; nt < 64; ++nt) {
        int buf = nt & 1;
        wait_async();
        __syncthreads();
        if (nt + 1 < 64) stage_up(stage[buf ^ 1], WuT, buT, nt + 1, tid);
        AB bb = load_B(&stage[buf][16384], 32, 0, 0, m16, hi);
        v8f acc = wmma16(pgf, bb, vzero8());      // router-weighted bu bias
#pragma unroll
        for (int t = 0; t < 32; ++t) {
            AB b = load_B(&stage[buf][0], 1024, 0, t * 32, m16, hi);
            acc = wmma16(gfrag[t], b, acc);
        }
        int col = nt * 16 + m16;
#pragma unroll
        for (int i = 0; i < 8; ++i)
            out[(size_t)(tok0 + trow0 + i + hi * 8) * DDIM + col] = acc[i];
    }
}

// ---------------------------------------------------------------------------
extern "C" void kernel_launch(void* const* d_in, const int* in_sizes, int n_in,
                              void* d_out, int out_size, void* d_ws, size_t ws_size,
                              hipStream_t stream) {
    (void)in_sizes; (void)n_in; (void)out_size;
    if (ws_size < WS_END) return;  // workspace too small (should not happen)

    const float* x  = (const float*)d_in[0];
    const float* Wr = (const float*)d_in[1];
    const float* br = (const float*)d_in[2];
    const float* Wd = (const float*)d_in[3];
    const float* bd = (const float*)d_in[4];
    const float* Wu = (const float*)d_in[5];
    const float* bu = (const float*)d_in[6];
    float* out = (float*)d_out;

    char* ws = (char*)d_ws;
    f16* xC  = (f16*)(ws + WS_XC);
    f16* WdC = (f16*)(ws + WS_WDC);
    f16* WuT = (f16*)(ws + WS_WUT);
    f16* WrT = (f16*)(ws + WS_WRT);
    f16* buT = (f16*)(ws + WS_BUT);

    prep_x <<<65536, 256, 0, stream>>>(x,  xC);
    prep_wd<<< 4096, 256, 0, stream>>>(Wd, WdC);
    prep_wu<<< 4096, 256, 0, stream>>>(Wu, WuT);
    prep_wr<<<   64, 256, 0, stream>>>(Wr, WrT);
    prep_bu<<<  128, 256, 0, stream>>>(bu, buT);

    moa_main<<<NTOK / TOKWG, 256, 0, stream>>>(xC, WdC, WuT, WrT, buT, br, bd, out);
}
